// st_encoder_20040317403592
// MI455X (gfx1250) — matrix operand order repaired and checked by reading, not compiled, not run on hardware
//
#include <hip/hip_runtime.h>
#include <hip/hip_bf16.h>

// Problem sizes
#define B_TOT  2048
#define T_LEN  128
#define CIN    2
#define COUT   16
#define H_REAL 232
#define HP     256          // padded hidden
#define GPAD   768          // 3*HP padded gate dim
#define NTILES 48           // GPAD/16
#define KCH    8            // HP/32 (bf16 WMMA K=32)
#define TILE_B 16
#define THREADS 256

typedef __attribute__((ext_vector_type(16))) __bf16 v16bf;
typedef __attribute__((ext_vector_type(8)))  float  v8f;

union Frag { unsigned u[8]; v16bf v; };

__device__ __forceinline__ unsigned short f2bf(float f) {
  union { float f; unsigned u; } x; x.f = f;
  unsigned u = x.u;
  u += 0x7FFFu + ((u >> 16) & 1u);   // round to nearest even
  return (unsigned short)(u >> 16);
}
__device__ __forceinline__ float bf2f(unsigned short h) {
  union { unsigned u; float f; } x; x.u = ((unsigned)h) << 16;
  return x.f;
}
__device__ __forceinline__ unsigned packbf(float a, float b) {
  return (unsigned)f2bf(a) | ((unsigned)f2bf(b) << 16);
}

// ---------------------------------------------------------------------------
// Prep: swizzle w_hh (f32 [696,232]) into bf16 B-fragment order, padded to
// [K=256][N=768]. Fragment block (nt,kc) = 256 dwords laid out [lane][vgpr].
// B k order used here (A matches): VGPR v, lanes0-15 -> K=2v,2v+1 ; lanes16-31 -> +16.
// ---------------------------------------------------------------------------
__global__ void pack_whh(const float* __restrict__ whh, unsigned* __restrict__ dst) {
  int gid = blockIdx.x * blockDim.x + threadIdx.x;
  if (gid >= NTILES * KCH * 256) return;
  int blk  = gid >> 8;
  int lane = (gid & 255) >> 3;
  int v    = gid & 7;
  int nt = blk >> 3;
  int kc = blk & 7;
  int k0 = kc * 32 + 2 * v + ((lane < 16) ? 0 : 16);
  int n  = lane & 15;
  int gp = nt * 16 + n;             // padded gate index
  int gate = gp >> 8;               // 0=r,1=z,2=n
  int idx  = gp & 255;
  bool vn = (idx < H_REAL);
  int g = gate * H_REAL + idx;
  float f0 = (vn && (k0     < H_REAL)) ? whh[g * H_REAL + k0]     : 0.f;
  float f1 = (vn && (k0 + 1 < H_REAL)) ? whh[g * H_REAL + k0 + 1] : 0.f;
  dst[gid] = packbf(f0, f1);
}

// w_ih (f32 [696,16]) -> bf16 B frags, K padded 16->32, N padded to 768.
__global__ void pack_wih(const float* __restrict__ wih, unsigned* __restrict__ dst) {
  int gid = blockIdx.x * blockDim.x + threadIdx.x;
  if (gid >= NTILES * 256) return;
  int nt   = gid >> 8;
  int lane = (gid & 255) >> 3;
  int v    = gid & 7;
  int k0 = 2 * v + ((lane < 16) ? 0 : 16);
  int n  = lane & 15;
  int gp = nt * 16 + n;
  int gate = gp >> 8;
  int idx  = gp & 255;
  bool vn = (idx < H_REAL);
  int g = gate * H_REAL + idx;
  float f0 = (vn && (k0     < COUT)) ? wih[g * COUT + k0]     : 0.f;
  float f1 = (vn && (k0 + 1 < COUT)) ? wih[g * COUT + k0 + 1] : 0.f;
  dst[gid] = packbf(f0, f1);
}

__global__ void pack_bias(const float* __restrict__ bih, const float* __restrict__ bhh,
                          float* __restrict__ dih, float* __restrict__ dhh) {
  int gid = blockIdx.x * blockDim.x + threadIdx.x;
  if (gid >= GPAD) return;
  int gate = gid >> 8, idx = gid & 255;
  float vi = (idx < H_REAL) ? bih[gate * H_REAL + idx] : 0.f;
  float vh = (idx < H_REAL) ? bhh[gate * H_REAL + idx] : 0.f;
  dih[gid] = vi; dhh[gid] = vh;
}

// ---------------------------------------------------------------------------
// Fused conv + GRU scan. One workgroup = 16 batch rows, full 128-step scan.
// Wave w owns gate-triplet columns q in {w, w+8}: N-tiles {q, 16+q, 32+q},
// so r/z/n pre-activations are element-aligned in its accumulators and the
// gate nonlinearity happens entirely in registers.
// ---------------------------------------------------------------------------
__global__ void __launch_bounds__(THREADS)
__attribute__((amdgpu_waves_per_eu(2)))
gru_scan(const float* __restrict__ X, const float* __restrict__ convw,
         const float* __restrict__ convb,
         const unsigned* __restrict__ whh_swz,
         const unsigned* __restrict__ wih_swz,
         const float* __restrict__ bih, const float* __restrict__ bhh,
         float* __restrict__ out)
{
  extern __shared__ char smem[];
  unsigned short* xe = (unsigned short*)smem;             // [16][128][16] 64KB
  unsigned short* hl = (unsigned short*)(smem + 65536);   // [16][256]      8KB

  const int tid = threadIdx.x;
  const int b0  = blockIdx.x * TILE_B;

  for (int i = tid; i < TILE_B * HP; i += THREADS) hl[i] = 0;

  // Conv1d(2->16,k=3,pad=1) + ReLU -> bf16 x_embed tile in LDS
  for (int o = tid; o < TILE_B * T_LEN * COUT; o += THREADS) {
    int c  = o & 15;
    int t  = (o >> 4) & 127;
    int bl = o >> 11;
    float acc = convb[c];
#pragma unroll
    for (int i = 0; i < CIN; ++i)
#pragma unroll
      for (int k = 0; k < 3; ++k) {
        int tt = t + k - 1;
        if (tt >= 0 && tt < T_LEN)
          acc += X[((b0 + bl) * T_LEN + tt) * CIN + i] * convw[(c * CIN + i) * 3 + k];
      }
    acc = acc > 0.f ? acc : 0.f;
    xe[(bl * T_LEN + t) * COUT + c] = f2bf(acc);
  }
  __syncthreads();

  const int wave = tid >> 5;
  const int lane = tid & 31;
  const int m    = lane & 15;
  const int hi   = (lane < 16) ? 0 : 8;   // A-frag / D-frag upper-half offset

  // w_ih fragments in registers: wif[g*2+qi] for nt = g*16 + wave + 8*qi
  Frag wif[6];
#pragma unroll
  for (int g = 0; g < 3; ++g)
#pragma unroll
    for (int qi = 0; qi < 2; ++qi) {
      int nt = g * 16 + wave + 8 * qi;
      const unsigned* p = wih_swz + nt * 256 + lane * 8;
#pragma unroll
      for (int v = 0; v < 8; ++v) wif[g * 2 + qi].u[v] = p[v];
    }

  // Per-lane folded bias constants for the two owned columns
  float cr[2], cz[2], cin_[2], chn[2];
#pragma unroll
  for (int qi = 0; qi < 2; ++qi) {
    int col = (wave + 8 * qi) * 16 + m;
    cr[qi]  = bih[col]          + bhh[col];
    cz[qi]  = bih[HP + col]     + bhh[HP + col];
    cin_[qi] = bih[2 * HP + col];
    chn[qi]  = bhh[2 * HP + col];
  }

  float hreg[2][8];
#pragma unroll
  for (int qi = 0; qi < 2; ++qi)
#pragma unroll
    for (int v = 0; v < 8; ++v) hreg[qi][v] = 0.f;

#pragma unroll 1
  for (int t = 0; t < T_LEN; ++t) {
    // A fragments of h (16-bit A 16x32 layout), all 8 K-chunks
    Frag af[KCH];
#pragma unroll
    for (int kc = 0; kc < KCH; ++kc) {
#pragma unroll
      for (int v = 0; v < 8; ++v) {
        int k0 = kc * 32 + 2 * v + 2 * hi;
        af[kc].u[v] = *(const unsigned*)&hl[m * HP + k0];
      }
    }
    // A fragment of x_embed[:,t,:] (K padded 16->32 with zeros)
    Frag ax;
#pragma unroll
    for (int v = 0; v < 8; ++v) {
      int k0 = 2 * v + 2 * hi;
      ax.u[v] = (k0 < COUT) ? *(const unsigned*)&xe[(m * T_LEN + t) * COUT + k0] : 0u;
    }
    __syncthreads();   // everyone has read old h

    // Launder a zero offset so B-frag loads can't be hoisted across t
    unsigned lz = 0;
    asm volatile("" : "+v"(lz));
    const unsigned* bb = whh_swz + lz;

#pragma unroll
    for (int qi = 0; qi < 2; ++qi) {
      int q = wave + 8 * qi;

      // r-gate: gi seeds the gh accumulation (i_r + h_r fused)
      v8f accr = __builtin_amdgcn_wmma_f32_16x16x32_bf16(
                     false, ax.v, false, wif[0 * 2 + qi].v, (short)0, (v8f){}, false, false);
      // z-gate
      v8f accz = __builtin_amdgcn_wmma_f32_16x16x32_bf16(
                     false, ax.v, false, wif[1 * 2 + qi].v, (short)0, (v8f){}, false, false);
      // n-gate: i and h parts must stay separate (n = tanh(i_n + r*h_n))
      v8f acni = __builtin_amdgcn_wmma_f32_16x16x32_bf16(
                     false, ax.v, false, wif[2 * 2 + qi].v, (short)0, (v8f){}, false, false);
      v8f acnh = {};

#pragma unroll
      for (int kc = 0; kc < KCH; ++kc) {
        Frag bf0, bf1, bf2;
        const unsigned* p0 = bb + ((q)      * KCH + kc) * 256 + lane * 8;
        const unsigned* p1 = bb + ((16 + q) * KCH + kc) * 256 + lane * 8;
        const unsigned* p2 = bb + ((32 + q) * KCH + kc) * 256 + lane * 8;
#pragma unroll
        for (int v = 0; v < 8; ++v) { bf0.u[v] = p0[v]; bf1.u[v] = p1[v]; bf2.u[v] = p2[v]; }
        accr = __builtin_amdgcn_wmma_f32_16x16x32_bf16(
                   false, af[kc].v, false, bf0.v, (short)0, accr, false, false);
        accz = __builtin_amdgcn_wmma_f32_16x16x32_bf16(
                   false, af[kc].v, false, bf1.v, (short)0, accz, false, false);
        acnh = __builtin_amdgcn_wmma_f32_16x16x32_bf16(
                   false, af[kc].v, false, bf2.v, (short)0, acnh, false, false);
      }

      // Gate nonlinearity in registers; write new h (bf16) for next step
      int col = q * 16 + m;
#pragma unroll
      for (int v = 0; v < 8; ++v) {
        float r = 1.f / (1.f + __expf(-(accr[v] + cr[qi])));
        float z = 1.f / (1.f + __expf(-(accz[v] + cz[qi])));
        float n = tanhf(acni[v] + cin_[qi] + r * (acnh[v] + chn[qi]));
        float h = (1.f - z) * n + z * hreg[qi][v];
        hreg[qi][v] = h;
        hl[(v + hi) * HP + col] = f2bf(h);
      }
    }
    __syncthreads();   // new h visible to all waves
  }

  // Emit final hidden state (f32, unpadded)
  for (int i = tid; i < TILE_B * H_REAL; i += THREADS) {
    int mm = i / H_REAL;
    int hp = i % H_REAL;
    out[(b0 + mm) * H_REAL + hp] = bf2f(hl[mm * HP + hp]);
  }
}

// ---------------------------------------------------------------------------
extern "C" void kernel_launch(void* const* d_in, const int* in_sizes, int n_in,
                              void* d_out, int out_size, void* d_ws, size_t ws_size,
                              hipStream_t stream) {
  (void)in_sizes; (void)n_in; (void)out_size; (void)ws_size;
  const float* X     = (const float*)d_in[0];
  const float* convw = (const float*)d_in[1];
  const float* convb = (const float*)d_in[2];
  const float* wih   = (const float*)d_in[3];
  const float* whh   = (const float*)d_in[4];
  const float* b_ih  = (const float*)d_in[5];
  const float* b_hh  = (const float*)d_in[6];
  float* out = (float*)d_out;

  // Workspace layout (bytes): whh_swz 384K | wih_swz 48K | bih 3K | bhh 3K
  unsigned* whh_swz = (unsigned*)d_ws;                  // 98304 dwords
  unsigned* wih_swz = whh_swz + NTILES * KCH * 256;     // 12288 dwords
  float* bih_pad = (float*)(wih_swz + NTILES * 256);    // 768 floats
  float* bhh_pad = bih_pad + GPAD;                      // 768 floats

  pack_whh<<<(NTILES * KCH * 256 + 255) / 256, 256, 0, stream>>>(whh, whh_swz);
  pack_wih<<<(NTILES * 256 + 255) / 256, 256, 0, stream>>>(wih, wih_swz);
  pack_bias<<<(GPAD + 255) / 256, 256, 0, stream>>>(b_ih, b_hh, bih_pad, bhh_pad);

  const size_t lds_bytes = 73728;   // 72 KB << 320 KB/WGP
  gru_scan<<<B_TOT / TILE_B, THREADS, lds_bytes, stream>>>(
      X, convw, convb, whh_swz, wih_swz, bih_pad, bhh_pad, out);
}
